// SGUBlock_46943992545483
// MI455X (gfx1250) — matrix operand order repaired and checked by reading, compiled, not run on hardware
//
#include <hip/hip_runtime.h>
#include <cstddef>

// ---------------- CDNA5 wave32 WMMA helpers ----------------
typedef _Float16 half8  __attribute__((ext_vector_type(8)));
typedef _Float16 half16 __attribute__((ext_vector_type(16)));
typedef float    float8 __attribute__((ext_vector_type(8)));

// A fragment: 16x32 f16, row-major source (pitch in halves, multiple of 8).
__device__ __forceinline__ half16 frag_a(const _Float16* base, int pitch, int lane) {
    int m    = lane & 15;
    int koff = (lane >> 4) * 8;
    const _Float16* p = base + m * pitch + koff;
    half8 lo = *(const half8*)(p);        // K = koff .. koff+7
    half8 hi = *(const half8*)(p + 16);   // K = koff+16 .. koff+23
    half16 r;
#pragma unroll
    for (int i = 0; i < 8; ++i) { r[i] = lo[i]; r[i + 8] = hi[i]; }
    return r;
}

// B fragment: 32x16 f16 from an N-major (transposed) tile: base[n*pitch + k].
__device__ __forceinline__ half16 frag_b(const _Float16* base, int pitch, int lane) {
    int n  = lane & 15;
    int kb = (lane >> 4) * 8;
    const _Float16* p = base + n * pitch + kb;
    half8 lo = *(const half8*)(p);
    half8 hi = *(const half8*)(p + 16);
    half16 r;
#pragma unroll
    for (int i = 0; i < 8; ++i) { r[i] = lo[i]; r[i + 8] = hi[i]; }
    return r;
}

__device__ __forceinline__ float8 wmma32(half16 a, half16 b, float8 c) {
    return __builtin_amdgcn_wmma_f32_16x16x32_f16(false, a, false, b, (short)0, c,
                                                  false, false);
}

__device__ __forceinline__ float gelu_exact(float x) {
    return 0.5f * x * (1.0f + erff(x * 0.70710678118654752f));
}

// ---------------- Problem constants ----------------
#define BB   512
#define SS   64
#define DD   128
#define BSR  (BB * SS)      // 32768 rows
#define DEXP 512
#define D3   1536
#define DATT 256
#define DSGU 512
#define TF   8192
#define HH   9011
#define HPAD 9024           // h3 pitch, 141*64, multiple of 32

// ---------------- Fused dual LayerNorm of x (D=128, 128 threads) ----------------
__global__ void ln_dual_kernel(const float* __restrict__ x,
                               const float* __restrict__ g1, const float* __restrict__ b1,
                               const float* __restrict__ g2, const float* __restrict__ b2,
                               _Float16* __restrict__ o1, _Float16* __restrict__ o2) {
    __shared__ float rs[128], rq[128];
    const int tid = threadIdx.x;
    const size_t row = blockIdx.x;
    float v = x[row * DD + tid];
    rs[tid] = v; rq[tid] = v * v;
    __syncthreads();
    for (int o = 64; o > 0; o >>= 1) {
        if (tid < o) { rs[tid] += rs[tid + o]; rq[tid] += rq[tid + o]; }
        __syncthreads();
    }
    float mean = rs[0] * (1.f / DD);
    float var  = rq[0] * (1.f / DD) - mean * mean;
    float nv   = (v - mean) * rsqrtf(var + 1e-5f);
    o1[row * DD + tid] = (_Float16)(nv * g1[tid] + b1[tid]);
    o2[row * DD + tid] = (_Float16)(nv * g2[tid] + b2[tid]);
}

// ---------------- Generic LayerNorm over last dim ----------------
template <int OUT16>
__global__ void ln_rows_kernel(const float* __restrict__ in, const float* __restrict__ g,
                               const float* __restrict__ be, void* __restrict__ out, int C) {
    __shared__ float rs[256], rq[256];
    const int tid = threadIdx.x, bd = blockDim.x;
    const size_t row = blockIdx.x;
    const float* x = in + row * (size_t)C;
    float s = 0.f, q = 0.f;
    for (int c = tid; c < C; c += bd) { float v = x[c]; s += v; q += v * v; }
    rs[tid] = s; rq[tid] = q;
    __syncthreads();
    for (int o = bd >> 1; o > 0; o >>= 1) {
        if (tid < o) { rs[tid] += rs[tid + o]; rq[tid] += rq[tid + o]; }
        __syncthreads();
    }
    float mean = rs[0] / (float)C;
    float var  = rq[0] / (float)C - mean * mean;
    float inv  = rsqrtf(var + 1e-5f);
    for (int c = tid; c < C; c += bd) {
        float v = (x[c] - mean) * inv * g[c] + be[c];
        if (OUT16) ((_Float16*)out)[row * (size_t)C + c] = (_Float16)v;
        else       ((float*)out)[row * (size_t)C + c] = v;
    }
}

// ---------------- Fused residual-add + LayerNorm over 8192 (MLP input) ----------------
__global__ void ln_add_rows_kernel(const float* __restrict__ x, const float* __restrict__ res,
                                   const float* __restrict__ g, const float* __restrict__ be,
                                   float* __restrict__ xnew, _Float16* __restrict__ lnf) {
    __shared__ float rs[256], rq[256];
    const int tid = threadIdx.x;
    const size_t row = blockIdx.x;
    const float* xr = x + row * (size_t)TF;
    const float* rr = res + row * (size_t)TF;
    float v[32];
    float s = 0.f, q = 0.f;
#pragma unroll
    for (int i = 0; i < 32; ++i) {
        int c = tid + i * 256;
        float t = xr[c] + rr[c];
        v[i] = t;
        xnew[row * (size_t)TF + c] = t;
        s += t; q += t * t;
    }
    rs[tid] = s; rq[tid] = q;
    __syncthreads();
    for (int o = 128; o > 0; o >>= 1) {
        if (tid < o) { rs[tid] += rs[tid + o]; rq[tid] += rq[tid + o]; }
        __syncthreads();
    }
    float mean = rs[0] * (1.f / TF);
    float var  = rq[0] * (1.f / TF) - mean * mean;
    float inv  = rsqrtf(var + 1e-5f);
#pragma unroll
    for (int i = 0; i < 32; ++i) {
        int c = tid + i * 256;
        lnf[row * (size_t)TF + c] = (_Float16)((v[i] - mean) * inv * g[c] + be[c]);
    }
}

// ---------------- Generic 64x64 tiled WMMA GEMM (L2-resident operands) ----------------
// Register-staged software pipeline: tile k+1 loads issue during tile k's WMMA phase.
template <int ACT, int OUT16>
__global__ void gemm_kernel(const _Float16* __restrict__ A, const float* __restrict__ W,
                            const float* __restrict__ bias, void* __restrict__ out,
                            int M, int N, int K, int lda, int ldw, int ldo) {
    __shared__ _Float16 As[64 * 40];   // [m][k], pitch 40
    __shared__ _Float16 Bs[64 * 40];   // transposed W tile: [n][k], pitch 40
    const int m0 = blockIdx.y * 64, n0 = blockIdx.x * 64;
    const int tid = threadIdx.x, lane = tid & 31, w = tid >> 5;
    const int rt = (w & 3) * 16, cb = (w >> 2) * 32;
    float8 acc0 = {}, acc1 = {};

    const int arow = tid >> 2, aj = (tid & 3) * 8;   // A tile mapping (coalesced half8)
    const int nloc = tid & 63, kb = tid >> 6;        // W tile mapping (coalesced f32)
    const int gm = m0 + arow, gn = n0 + nloc;

    half8 areg;
    float wreg[8];
    auto load_tile = [&](int k0) {
        half8 za = {};
        if (gm < M) {
            if (((lda & 7) == 0) && (k0 + 32 <= K)) {
                za = *(const half8*)(A + (size_t)gm * lda + k0 + aj);
            } else {
#pragma unroll
                for (int e = 0; e < 8; ++e) {
                    int gk = k0 + aj + e;
                    za[e] = (gk < K) ? A[(size_t)gm * lda + gk] : (_Float16)0.f;
                }
            }
        }
        areg = za;
#pragma unroll
        for (int i = 0; i < 8; ++i) {
            int gk = k0 + kb + i * 4;
            wreg[i] = (gk < K && gn < N) ? W[(size_t)gk * ldw + gn] : 0.f;
        }
    };

    load_tile(0);
    for (int k0 = 0; k0 < K; k0 += 32) {
        *(half8*)(As + arow * 40 + aj) = areg;
#pragma unroll
        for (int i = 0; i < 8; ++i) Bs[nloc * 40 + kb + i * 4] = (_Float16)wreg[i];
        __syncthreads();
        if (k0 + 32 < K) load_tile(k0 + 32);   // overlap next loads with compute
        half16 a  = frag_a(As + rt * 40, 40, lane);
        half16 b0 = frag_b(Bs + cb * 40, 40, lane);
        half16 b1 = frag_b(Bs + (cb + 16) * 40, 40, lane);
        acc0 = wmma32(a, b0, acc0);
        acc1 = wmma32(a, b1, acc1);
        __syncthreads();
    }

    const int hi8 = (lane >> 4) * 8, nn = lane & 15;
#pragma unroll
    for (int r = 0; r < 8; ++r) {
        int gmo = m0 + rt + r + hi8;
        if (gmo >= M) continue;
        int gn0 = n0 + cb + nn;
        int gn1 = gn0 + 16;
        float v0 = acc0[r] + (bias ? bias[gn0 < N ? gn0 : 0] : 0.f);
        float v1 = acc1[r] + (bias ? bias[gn1 < N ? gn1 : 0] : 0.f);
        if (ACT) { v0 = gelu_exact(v0); v1 = gelu_exact(v1); }
        if (gn0 < N) {
            if (OUT16) ((_Float16*)out)[(size_t)gmo * ldo + gn0] = (_Float16)v0;
            else       ((float*)out)[(size_t)gmo * ldo + gn0] = v0;
        }
        if (gn1 < N) {
            if (OUT16) ((_Float16*)out)[(size_t)gmo * ldo + gn1] = (_Float16)v1;
            else       ((float*)out)[(size_t)gmo * ldo + gn1] = v1;
        }
    }
}

// ---------------- Tall weight-streaming WMMA GEMM: block = [512 x 64] ----------------
// Weights (HBM-resident, > L2) are read exactly once, non-temporally; A (f16, L2-
// resident) is re-read per N-block. Register-staged double buffering hides HBM latency.
// Ka = padded K (multiple of 32, A readable up to Ka); W guarded by true K.
template <int ACT, int OUT16>
__global__ void gemm_tall_kernel(const _Float16* __restrict__ A, const float* __restrict__ W,
                                 const float* __restrict__ bias, void* __restrict__ out,
                                 int N, int K, int Ka, int lda, int ldw, int ldo) {
    __shared__ _Float16 As[512 * 40];  // 512x32 tile, 40 KB
    __shared__ _Float16 Bs[64 * 40];   // transposed W tile [n][k], 5 KB
    const int n0 = blockIdx.x * 64;
    const int tid = threadIdx.x, lane = tid & 31, w = tid >> 5;
    const int rbase = w * 64;                        // wave's 64 rows
    const int arow = tid >> 2, aj = (tid & 3) * 8;   // A tile mapping
    const int nloc = tid & 63, kb = tid >> 6;        // W tile mapping
    const int gn = n0 + nloc;

    float8 acc[4][4] = {};
    half8 areg[8];
    float wreg[8];

    auto load_tile = [&](int k0) {
#pragma unroll
        for (int i = 0; i < 8; ++i) {
            int r = arow + i * 64;
            areg[i] = *(const half8*)(A + (size_t)r * lda + k0 + aj);
        }
#pragma unroll
        for (int i = 0; i < 8; ++i) {
            int gk = k0 + kb + i * 4;
            wreg[i] = (gk < K && gn < N)
                          ? __builtin_nontemporal_load(&W[(size_t)gk * ldw + gn])
                          : 0.f;
        }
        int pk = k0 + 64 + kb;   // prime L2 one tile past the register stage
        if (pk < K && gn < N) __builtin_prefetch(&W[(size_t)pk * ldw + gn], 0, 1);
    };

    load_tile(0);
    for (int k0 = 0; k0 < Ka; k0 += 32) {
#pragma unroll
        for (int i = 0; i < 8; ++i)
            *(half8*)(As + (arow + i * 64) * 40 + aj) = areg[i];
#pragma unroll
        for (int i = 0; i < 8; ++i) Bs[nloc * 40 + kb + i * 4] = (_Float16)wreg[i];
        __syncthreads();
        if (k0 + 32 < Ka) load_tile(k0 + 32);   // overlap next loads with 16 WMMAs
        half16 afr[4];
#pragma unroll
        for (int s = 0; s < 4; ++s) afr[s] = frag_a(As + (rbase + s * 16) * 40, 40, lane);
#pragma unroll
        for (int c = 0; c < 4; ++c) {
            half16 bf = frag_b(Bs + c * 16 * 40, 40, lane);
#pragma unroll
            for (int s = 0; s < 4; ++s) acc[s][c] = wmma32(afr[s], bf, acc[s][c]);
        }
        __syncthreads();
    }

    const int hi8 = (lane >> 4) * 8, nn = lane & 15;
#pragma unroll
    for (int s = 0; s < 4; ++s) {
#pragma unroll
        for (int c = 0; c < 4; ++c) {
#pragma unroll
            for (int r = 0; r < 8; ++r) {
                int gm  = rbase + s * 16 + r + hi8;
                int gnc = n0 + c * 16 + nn;
                float v = 0.f;   // pad columns (N..ldo) get zeros (h3 pitch padding)
                if (gnc < N) {
                    v = acc[s][c][r] + (bias ? bias[gnc] : 0.f);
                    if (ACT) v = gelu_exact(v);
                }
                if (OUT16) ((_Float16*)out)[(size_t)gm * ldo + gnc] = (_Float16)v;
                else if (gnc < N) ((float*)out)[(size_t)gm * ldo + gnc] = v;
            }
        }
    }
}

// ---------------- Fused per-batch tiny attention ----------------
__global__ void attn_kernel(const _Float16* __restrict__ qkvh,
                            const float* __restrict__ rel_bias,
                            const float* __restrict__ scale_p,
                            _Float16* __restrict__ av_out) {
    extern __shared__ char smem[];
    _Float16* qs  = (_Float16*)smem;             // [64][512]
    _Float16* ksm = qs + 64 * 512;               // [64][512]
    _Float16* vsT = ksm + 64 * 512;              // [512][64]  (v transposed)
    float*    sc  = (float*)(vsT + 512 * 64);    // [64][64]
    _Float16* p16 = (_Float16*)(sc + 64 * 64);   // [64][72]

    const int b = blockIdx.x, tid = threadIdx.x;
    const _Float16* base = qkvh + (size_t)b * 64 * 1536;
    for (int vi = tid; vi < 64 * 64; vi += 256) {   // half8-vectorized staging
        int s = vi >> 6, d8 = (vi & 63) * 8;
        *(half8*)(qs + s * 512 + d8)  = *(const half8*)(base + s * 1536 + d8);
        *(half8*)(ksm + s * 512 + d8) = *(const half8*)(base + s * 1536 + 512 + d8);
        half8 vv = *(const half8*)(base + s * 1536 + 1024 + d8);
#pragma unroll
        for (int e = 0; e < 8; ++e) vsT[(d8 + e) * 64 + s] = vv[e];
    }
    __syncthreads();

    const int lane = tid & 31, w = tid >> 5;
    const int hi8 = (lane >> 4) * 8, nn = lane & 15;
    {   // S = q @ k^T : 64x64, K=512
        int rt = (w & 3) * 16, cb = (w >> 2) * 32;
        float8 acc0 = {}, acc1 = {};
        for (int kk = 0; kk < 512; kk += 32) {
            half16 a  = frag_a(qs + rt * 512 + kk, 512, lane);
            half16 b0 = frag_b(ksm + cb * 512 + kk, 512, lane);
            half16 b1 = frag_b(ksm + (cb + 16) * 512 + kk, 512, lane);
            acc0 = wmma32(a, b0, acc0);
            acc1 = wmma32(a, b1, acc1);
        }
#pragma unroll
        for (int r = 0; r < 8; ++r) {
            int Mi = rt + r + hi8;
            int N0 = cb + nn, N1 = cb + 16 + nn;
            sc[Mi * 64 + N0] = acc0[r] + rel_bias[N0 - Mi + 63];
            sc[Mi * 64 + N1] = acc1[r] + rel_bias[N1 - Mi + 63];
        }
    }
    __syncthreads();

    {   // softmax row-wise, then * scale (post-softmax per reference)
        float scale = *scale_p;
        if (tid < 64) {
            float mx = -1e30f;
            for (int j = 0; j < 64; ++j) mx = fmaxf(mx, sc[tid * 64 + j]);
            float sum = 0.f;
            for (int j = 0; j < 64; ++j) sum += __expf(sc[tid * 64 + j] - mx);
            float inv = scale / sum;
            for (int j = 0; j < 64; ++j)
                p16[tid * 72 + j] = (_Float16)(__expf(sc[tid * 64 + j] - mx) * inv);
        }
    }
    __syncthreads();

    {   // AV = P[64,64] @ V[64,512]
        int rt = (w & 3) * 16;
        half16 a0 = frag_a(p16 + rt * 72 + 0, 72, lane);
        half16 a1 = frag_a(p16 + rt * 72 + 32, 72, lane);
        int cbase = (w >> 2) * 256;
        for (int cs = 0; cs < 16; ++cs) {
            int ct = cbase + cs * 16;
            float8 acc = {};
            acc = wmma32(a0, frag_b(vsT + ct * 64 + 0, 64, lane), acc);
            acc = wmma32(a1, frag_b(vsT + ct * 64 + 32, 64, lane), acc);
#pragma unroll
            for (int r = 0; r < 8; ++r) {
                int s = rt + r + hi8;
                av_out[((size_t)b * 64 + s) * 512 + ct + nn] = (_Float16)acc[r];
            }
        }
    }
}

// ---------------- Fused SGU spatial gate ----------------
__global__ void sgu_kernel(const float* __restrict__ pin, const float* __restrict__ a,
                           const float* __restrict__ sp_g, const float* __restrict__ sp_bln,
                           const float* __restrict__ sp_w, const float* __restrict__ sp_bias,
                           float* __restrict__ gated) {
    extern __shared__ char smem[];
    float*    ybuf = (float*)smem;                 // [256][68]  vt (then Y)
    _Float16* vt16 = (_Float16*)(ybuf + 256 * 68); // [256][72]  LN'd vt f16
    _Float16* bw   = vt16 + 256 * 72;              // [64][72]   sp_w transposed

    const int b = blockIdx.x, tid = threadIdx.x;
    const float* pbase = pin + (size_t)b * 64 * 512;
    for (int vi = tid; vi < 64 * 64; vi += 256) {   // vv -> vt (transposed), float4 loads
        int s = vi >> 6, c0 = (vi & 63) * 4;
        float4 t = *(const float4*)(pbase + s * 512 + 256 + c0);
        ybuf[(c0 + 0) * 68 + s] = t.x;
        ybuf[(c0 + 1) * 68 + s] = t.y;
        ybuf[(c0 + 2) * 68 + s] = t.z;
        ybuf[(c0 + 3) * 68 + s] = t.w;
    }
    for (int i = tid; i < 64 * 64; i += 256) {      // sp_w -> transposed f16
        int s = i >> 6, n = i & 63;
        bw[n * 72 + s] = (_Float16)sp_w[s * 64 + n];
    }
    __syncthreads();

    {   // LN over seq (64) per channel row
        int c = tid;
        float m = 0.f;
        for (int s = 0; s < 64; ++s) m += ybuf[c * 68 + s];
        m *= (1.f / 64.f);
        float v = 0.f;
        for (int s = 0; s < 64; ++s) { float d = ybuf[c * 68 + s] - m; v += d * d; }
        v *= (1.f / 64.f);
        float inv = rsqrtf(v + 1e-5f);
        for (int s = 0; s < 64; ++s)
            vt16[c * 72 + s] = (_Float16)((ybuf[c * 68 + s] - m) * inv * sp_g[s] + sp_bln[s]);
    }
    __syncthreads();

    const int lane = tid & 31, w = tid >> 5;
    const int hi8 = (lane >> 4) * 8, nn = lane & 15;
    for (int rs = 0; rs < 2; ++rs) {   // Y[256,64] = vt16 @ sp_w + b
        int rt = w * 32 + rs * 16;
        half16 a0 = frag_a(vt16 + rt * 72 + 0, 72, lane);
        half16 a1 = frag_a(vt16 + rt * 72 + 32, 72, lane);
        for (int ct4 = 0; ct4 < 4; ++ct4) {
            int ct = ct4 * 16;
            float8 acc = {};
            acc = wmma32(a0, frag_b(bw + ct * 72 + 0, 72, lane), acc);
            acc = wmma32(a1, frag_b(bw + ct * 72 + 32, 72, lane), acc);
#pragma unroll
            for (int r = 0; r < 8; ++r) {
                int c = rt + r + hi8;
                int sp = ct + nn;
                ybuf[c * 68 + sp] = acc[r] + sp_bias[sp];
            }
        }
    }
    __syncthreads();

    const float* abase = a + (size_t)b * 64 * 256;
    for (int vi = tid; vi < 64 * 64; vi += 256) {   // out = u * (gelu(Y)^T + a), float4
        int sp = vi >> 6, c0 = (vi & 63) * 4;
        float4 u  = *(const float4*)(pbase + sp * 512 + c0);
        float4 av = *(const float4*)(abase + sp * 256 + c0);
        float4 o;
        o.x = u.x * (gelu_exact(ybuf[(c0 + 0) * 68 + sp]) + av.x);
        o.y = u.y * (gelu_exact(ybuf[(c0 + 1) * 68 + sp]) + av.y);
        o.z = u.z * (gelu_exact(ybuf[(c0 + 2) * 68 + sp]) + av.z);
        o.w = u.w * (gelu_exact(ybuf[(c0 + 3) * 68 + sp]) + av.w);
        *(float4*)(gated + ((size_t)b * 64 + sp) * 256 + c0) = o;
    }
}

// ---------------- elementwise residual add ----------------
__global__ void add2_kernel(const float* __restrict__ a, const float* __restrict__ b,
                            float* __restrict__ o, int n) {
    int i = blockIdx.x * blockDim.x + threadIdx.x;
    if (i < n) o[i] = a[i] + b[i];
}

// ---------------- host orchestration ----------------
extern "C" void kernel_launch(void* const* d_in, const int* in_sizes, int n_in,
                              void* d_out, int out_size, void* d_ws, size_t ws_size,
                              hipStream_t stream) {
    (void)in_sizes; (void)n_in; (void)out_size; (void)ws_size;
    const float* x         = (const float*)d_in[0];
    const float* qkv_ln_g  = (const float*)d_in[1];
    const float* qkv_ln_b  = (const float*)d_in[2];
    const float* qkv_w     = (const float*)d_in[3];
    const float* qkv_b     = (const float*)d_in[4];
    const float* aproj_w   = (const float*)d_in[5];
    const float* aproj_b   = (const float*)d_in[6];
    const float* attn_ln_g = (const float*)d_in[7];
    const float* attn_ln_b = (const float*)d_in[8];
    const float* rel_bias  = (const float*)d_in[9];
    const float* scale     = (const float*)d_in[10];
    const float* pin_ln_g  = (const float*)d_in[11];
    const float* pin_ln_b  = (const float*)d_in[12];
    const float* pin_w     = (const float*)d_in[13];
    const float* pin_b     = (const float*)d_in[14];
    const float* sp_ln_g   = (const float*)d_in[15];
    const float* sp_ln_b   = (const float*)d_in[16];
    const float* sp_w      = (const float*)d_in[17];
    const float* sp_b      = (const float*)d_in[18];
    const float* pout_ln_g = (const float*)d_in[19];
    const float* pout_ln_b = (const float*)d_in[20];
    const float* pout_w    = (const float*)d_in[21];
    const float* pout_b    = (const float*)d_in[22];
    const float* mlp_ln_g  = (const float*)d_in[23];
    const float* mlp_ln_b  = (const float*)d_in[24];
    const float* mlp_w1    = (const float*)d_in[25];
    const float* mlp_b1    = (const float*)d_in[26];
    const float* mlp_w2    = (const float*)d_in[27];
    const float* mlp_b2    = (const float*)d_in[28];

    char* ws = (char*)d_ws;
    const size_t off_lnq  = 0;
    const size_t off_lnp  = off_lnq + (size_t)BSR * DD * 2;
    const size_t off_qkv  = off_lnp + (size_t)BSR * DD * 2;
    const size_t off_pin  = off_qkv + (size_t)BSR * D3 * 2;
    const size_t off_av   = off_pin + (size_t)BSR * DSGU * 4;
    const size_t off_a    = off_av + (size_t)BSR * DEXP * 2;
    // reuse of dead regions:
    const size_t off_gated = off_qkv;                            // qkv dead after attn
    const size_t off_lng   = off_gated + (size_t)BSR * DATT * 4;
    const size_t off_pout  = off_lng + (size_t)BSR * DATT * 2;
    const size_t off_xnew  = off_pout + (size_t)BSR * DD * 4;
    const size_t off_lnf   = off_lnq;                            // lnq dead after qkv gemm
    const size_t off_h3    = off_lnp;                            // lnp dead after pin gemm
    const size_t off_m2    = off_a;                              // a dead after sgu

    _Float16* lnq   = (_Float16*)(ws + off_lnq);
    _Float16* lnp   = (_Float16*)(ws + off_lnp);
    _Float16* qkvh  = (_Float16*)(ws + off_qkv);
    float*    pinb  = (float*)(ws + off_pin);
    _Float16* av    = (_Float16*)(ws + off_av);
    float*    abuf  = (float*)(ws + off_a);
    float*    gated = (float*)(ws + off_gated);
    _Float16* lng   = (_Float16*)(ws + off_lng);
    float*    poutb = (float*)(ws + off_pout);
    float*    xnew  = (float*)(ws + off_xnew);
    _Float16* lnf   = (_Float16*)(ws + off_lnf);
    _Float16* h3    = (_Float16*)(ws + off_h3);   // pitch HPAD, pad cols zeroed
    float*    m2    = (float*)(ws + off_m2);

    const int attn_lds = 64 * 512 * 2 * 2 + 512 * 64 * 2 + 64 * 64 * 4 + 64 * 72 * 2;
    const int sgu_lds  = 256 * 68 * 4 + 256 * 72 * 2 + 64 * 72 * 2;
    (void)hipFuncSetAttribute((const void*)attn_kernel,
                              hipFuncAttributeMaxDynamicSharedMemorySize, attn_lds);
    (void)hipFuncSetAttribute((const void*)sgu_kernel,
                              hipFuncAttributeMaxDynamicSharedMemorySize, sgu_lds);

    // 1. fused dual LN of x (qkv branch + SGU branch share mean/var)
    ln_dual_kernel<<<BSR, 128, 0, stream>>>(x, qkv_ln_g, qkv_ln_b, pin_ln_g, pin_ln_b,
                                            lnq, lnp);
    // 2. qkv projection  [BS,128]@[128,1536]
    gemm_kernel<0, 1><<<dim3(D3 / 64, BSR / 64), 256, 0, stream>>>(
        lnq, qkv_w, qkv_b, qkvh, BSR, D3, DD, DD, D3, D3);
    // 3. SGU proj_in + gelu  [BS,128]@[128,512]
    gemm_kernel<1, 0><<<dim3(DSGU / 64, BSR / 64), 256, 0, stream>>>(
        lnp, pin_w, pin_b, pinb, BSR, DSGU, DD, DD, DSGU, DSGU);
    // 4. tiny attention per batch
    attn_kernel<<<BB, 256, attn_lds, stream>>>(qkvh, rel_bias, scale, av);
    // 5. attention projection [BS,512]@[512,256]
    gemm_kernel<0, 0><<<dim3(DATT / 64, BSR / 64), 256, 0, stream>>>(
        av, aproj_w, aproj_b, abuf, BSR, DATT, DEXP, DEXP, DATT, DATT);
    // 6. LN -> a (in place)
    ln_rows_kernel<0><<<BSR, 256, 0, stream>>>(abuf, attn_ln_g, attn_ln_b, abuf, DATT);
    // 7. SGU spatial gating per batch
    sgu_kernel<<<BB, 256, sgu_lds, stream>>>(pinb, abuf, sp_ln_g, sp_ln_b, sp_w, sp_b, gated);
    // 8. LN of gated
    ln_rows_kernel<1><<<BSR, 256, 0, stream>>>(gated, pout_ln_g, pout_ln_b, lng, DATT);
    // 9. proj_out [BS,256]@[256,128]
    gemm_kernel<0, 0><<<dim3(DD / 64, BSR / 64), 256, 0, stream>>>(
        lng, pout_w, pout_b, poutb, BSR, DD, DATT, DATT, DD, DD);
    // 10. fused residual + MLP LayerNorm: xnew = x + out; lnf = LN(xnew)
    ln_add_rows_kernel<<<BB, 256, 0, stream>>>(x, poutb, mlp_ln_g, mlp_ln_b, xnew, lnf);
    // 11. MLP up + gelu: weight-streaming tall GEMM, h3 pitch HPAD (pad zeroed)
    gemm_tall_kernel<1, 1><<<HPAD / 64, 256, 0, stream>>>(
        lnf, mlp_w1, mlp_b1, h3, HH, TF, TF, TF, HH, HPAD);
    // 12. MLP down: tall GEMM, A reads padded h3 (Ka = HPAD)
    gemm_tall_kernel<0, 0><<<TF / 64, 256, 0, stream>>>(
        h3, mlp_w2, mlp_b2, m2, TF, HH, HPAD, HPAD, TF, TF);
    // 13. final residual -> d_out
    {
        int n = BB * TF;
        add2_kernel<<<(n + 255) / 256, 256, 0, stream>>>(xnew, m2, (float*)d_out, n);
    }
}
